// HMTextCrossAttention_15522011808289
// MI455X (gfx1250) — matrix operand order repaired and checked by reading, not compiled, not run on hardware
//
#include <hip/hip_runtime.h>
#include <hip/hip_bf16.h>

typedef __attribute__((ext_vector_type(16))) _Float16 v16h;
typedef __attribute__((ext_vector_type(8)))  float    v8f;
typedef _Float16 half_t;

#define NEG_INF_SCORE (-1.0e9f)

__device__ __forceinline__ v8f wmma_f16(v16h a, v16h b, v8f c) {
  // D = A(16x32 f16) * B(32x16 f16) + C(16x16 f32)
  return __builtin_amdgcn_wmma_f32_16x16x32_f16(
      /*neg_a=*/false, a, /*neg_b=*/false, b,
      /*c_mod=*/(short)0, c, /*reuse_a=*/false, /*reuse_b=*/false);
}

// --------------------------- TDM (Tensor Data Mover) -----------------------
#if defined(__has_builtin)
#if __has_builtin(__builtin_amdgcn_tensor_load_to_lds) && \
    __has_builtin(__builtin_amdgcn_s_wait_tensorcnt)
#define USE_TDM 1
#endif
#endif
#ifndef USE_TDM
#define USE_TDM 0
#endif

#if USE_TDM
typedef unsigned int u32x4 __attribute__((ext_vector_type(4)));
typedef int i32x8 __attribute__((ext_vector_type(8)));
typedef int i32x4 __attribute__((ext_vector_type(4)));

// 1-D contiguous tile copy: nelem f16 elements, global -> LDS.
// D# layout per CDNA5 ISA (groups 0/1; groups 2/3 zero).
__device__ __forceinline__ void tdm_load_1d_f16(const half_t* gsrc,
                                                unsigned lds_byte_off,
                                                unsigned nelem) {
  const unsigned long long ga = (unsigned long long)(uintptr_t)gsrc;
  u32x4 g0;
  g0[0] = 1u;                                    // count=1 (valid user D#)
  g0[1] = lds_byte_off;                          // lds_addr (bytes)
  g0[2] = (unsigned)(ga & 0xffffffffu);          // global_addr[31:0]
  g0[3] = (unsigned)((ga >> 32) & 0x01ffffffu)   // global_addr[56:32]
        | (2u << 30);                            // type = 2 ("image")
  const unsigned td0 = 0x40000000u;              // huge tensor_dim0: no clip
  i32x8 g1;
  g1[0] = (int)(1u << 16);                       // wg_mask=0, data_size=2B
  g1[1] = (int)((td0 & 0xffffu) << 16);          // tensor_dim0[15:0]
  g1[2] = (int)(((td0 >> 16) & 0xffffu)          // tensor_dim0[31:16]
        | (1u << 16));                           // tensor_dim1[15:0] = 1
  g1[3] = (int)((nelem & 0xffffu) << 16);        // tile_dim0 = nelem
  g1[4] = 0;                                     // tile_dim1=0, tile_dim2=0
  g1[5] = (int)nelem;                            // tensor_dim0_stride lo
  g1[6] = 0;
  g1[7] = 0;
  i32x4 z4 = {0, 0, 0, 0};
#if defined(__clang_major__) && (__clang_major__ >= 23)
  // clang-23 / therock-10.0: 6-arg form (extra int32x8 before cpol)
  i32x8 z8 = {0, 0, 0, 0, 0, 0, 0, 0};
  __builtin_amdgcn_tensor_load_to_lds(g0, g1, z4, z4, z8, 0);
#else
  // ROCm 7.2 clang-22: 5-arg form
  __builtin_amdgcn_tensor_load_to_lds(g0, g1, z4, z4, 0);
#endif
}
#endif

// ---------------------------------------------------------------------------
// Pass 1: Y[M,1024] (f16) = f16( X[M,1024](f32) @ W[1024,1024](f32) + bias )
// WG = 128 threads (4 waves). 64x64 output tile, K-step 32.
// ---------------------------------------------------------------------------
#define GD 1024
#define TM 64
#define TN 64
#define TK 32

__global__ __launch_bounds__(128) void proj_gemm_f16(
    const float* __restrict__ X, const float* __restrict__ W,
    const float* __restrict__ bias, half_t* __restrict__ Y, int M) {
  __shared__ half_t As[TM][TK];   // X tile, row-major [m][k]
  __shared__ half_t Bt[TN][TK];   // W tile transposed: Bt[n][k] = W[k0+k][n0+n]

  const int m0 = blockIdx.x * TM;
  const int n0 = blockIdx.y * TN;
  const int tid  = threadIdx.x;
  const int wave = tid >> 5;
  const int lane = tid & 31;

  v8f acc[4];
  for (int t = 0; t < 4; ++t) acc[t] = (v8f){0,0,0,0,0,0,0,0};

  for (int k0 = 0; k0 < GD; k0 += TK) {
    // stage X tile: 4 straight-line B128 loads/thread, packed f16 B64 stores
#pragma unroll
    for (int j = 0; j < 4; ++j) {
      const int i = tid + j * 128;           // compile-time trip count
      const int r = i >> 3, c4 = (i & 7) * 4;
      const float4 x = *(const float4*)&X[(size_t)(m0 + r) * GD + k0 + c4];
      half_t h[4] = {(half_t)x.x, (half_t)x.y, (half_t)x.z, (half_t)x.w};
      *(uint2*)&As[r][c4] = *(const uint2*)h;
    }
    // stage W tile transposed: 4 B128 loads/thread, b16 scatter stores
#pragma unroll
    for (int j = 0; j < 4; ++j) {
      const int i = tid + j * 128;
      const int k = i >> 4, n4 = (i & 15) * 4;
      const float4 w = *(const float4*)&W[(size_t)(k0 + k) * GD + n0 + n4];
      Bt[n4 + 0][k] = (half_t)w.x;
      Bt[n4 + 1][k] = (half_t)w.y;
      Bt[n4 + 2][k] = (half_t)w.z;
      Bt[n4 + 3][k] = (half_t)w.w;
    }
    __syncthreads();

    // A fragment (16x32): lanes 0-15 hold M=lane K={0..7,16..23};
    // lanes 16-31 hold M=lane-16, K={8..15,24..31}  (ISA 7.12.2)
    const int am = wave * 16 + (lane & 15);
    const int ks = (lane >> 4) * 8;
    v16h afrag;
    ((uint4*)&afrag)[0] = *(const uint4*)&As[am][ks];
    ((uint4*)&afrag)[1] = *(const uint4*)&As[am][16 + ks];

#pragma unroll
    for (int t = 0; t < 4; ++t) {
      // B fragment (32x16): lane n = lane&15, K = {kb..kb+7, 16+kb..16+kb+7}, kb=(lane>>4)*8
      const int bn = t * 16 + (lane & 15);
      const int kb = (lane >> 4) * 8;
      v16h bfrag;
      ((uint4*)&bfrag)[0] = *(const uint4*)&Bt[bn][kb];
      ((uint4*)&bfrag)[1] = *(const uint4*)&Bt[bn][kb + 16];
      acc[t] = wmma_f16(afrag, bfrag, acc[t]);
    }
    __syncthreads();
  }

  // epilogue: + bias, f32 -> f16 store. C layout: VGPR r -> row r (lanes 0-15)
  // or row r+8 (lanes 16-31); N = lane&15 within the 16-wide subtile.
  const int hsel = lane >> 4;
  const int nloc = lane & 15;
  const int mbase = m0 + wave * 16 + hsel * 8;
#pragma unroll
  for (int t = 0; t < 4; ++t) {
    const int col = n0 + t * 16 + nloc;
    const float bv = bias[col];
#pragma unroll
    for (int r = 0; r < 8; ++r) {
      Y[(size_t)(mbase + r) * GD + col] = (half_t)(acc[t][r] + bv);
    }
  }
}

// ---------------------------------------------------------------------------
// Pass 2: fused flash attention.
// WG = 128 threads (4 waves) owns (batch b, 16 query rows).
// Wave w owns d-chunk [256w, 256w+256) for BOTH the QK^T contraction
// (partial S, reduced via LDS) and the O accumulator (16x256 f32).
// Q fragments stay resident in VGPRs across the whole Lk loop.
// K tiles are streamed L2 -> LDS by the TDM engine, double-buffered.
// ---------------------------------------------------------------------------
#define LKT 32  // K/V tile along Lk

// stage V tile transposed: each thread handles a PAIR of adjacent Lk rows so
// the two transposed elements pack into one ds_store_b32.
__device__ __forceinline__ void stage_v_transposed(const half_t* __restrict__ Vb,
                                                   half_t* __restrict__ Vt,
                                                   int k0, int tid) {
#pragma unroll 4
  for (int j = 0; j < 16; ++j) {
    const int i = tid + j * 128;      // 2048 row-pair chunks
    const int r2 = (i >> 7) << 1;     // even Lk row of the pair
    const int c8 = i & 127;           // 8-half chunk within the row
    const uint4 v0 = *(const uint4*)(Vb + (size_t)(k0 + r2) * GD + c8 * 8);
    const uint4 v1 = *(const uint4*)(Vb + (size_t)(k0 + r2 + 1) * GD + c8 * 8);
    const half_t* h0 = (const half_t*)&v0;
    const half_t* h1 = (const half_t*)&v1;
#pragma unroll
    for (int e = 0; e < 8; ++e) {
      half_t pair[2] = {h0[e], h1[e]};
      *(unsigned*)&Vt[(c8 * 8 + e) * LKT + r2] = *(const unsigned*)pair;
    }
  }
}

__global__ __launch_bounds__(128) void flash_attn(
    const half_t* __restrict__ Q, const half_t* __restrict__ K,
    const half_t* __restrict__ V, const int* __restrict__ mask,
    float* __restrict__ O, int B, int Lq, int Lk) {
  extern __shared__ char smem[];
  half_t* Ks0 = (half_t*)smem;            // [32][1024] K tile buffer 0
  half_t* Ks1 = Ks0 + LKT * GD;           // [32][1024] K tile buffer 1 (TDM)
  half_t* Vt  = Ks1 + LKT * GD;           // [1024][32] V tile, transposed
  float*  Sp  = (float*)(Vt + LKT * GD);  // [4][16*32] partial S
  half_t* Pb  = (half_t*)(Sp + 4 * 16 * LKT);  // [16][32] P (f16)

  const int b    = blockIdx.y;
  const int q0   = blockIdx.x * 16;
  const int tid  = threadIdx.x;
  const int wave = tid >> 5;
  const int lane = tid & 31;
  const int chunk = wave * 256;  // this wave's d-range

  const half_t* Qb = Q + ((size_t)b * Lq + q0) * GD;
  const half_t* Kb = K + (size_t)b * Lk * GD;
  const half_t* Vb = V + (size_t)b * Lk * GD;
  const int* maskb = mask + (size_t)b * Lk;

  // ---- Q-stationary: preload 8 A-fragments (16 x 256 of Q) straight from
  // global (L2-resident after pass 1). A-fragment layout per ISA 7.12.2.
  const int qm = lane & 15;
  const int qks = (lane >> 4) * 8;
  v16h qf[8];
  {
    const half_t* qrow = Qb + (size_t)qm * GD + chunk;
#pragma unroll
    for (int s = 0; s < 8; ++s) {
      ((uint4*)&qf[s])[0] = *(const uint4*)(qrow + s * 32 + qks);
      ((uint4*)&qf[s])[1] = *(const uint4*)(qrow + s * 32 + 16 + qks);
    }
  }

  v8f o[16];
  for (int t = 0; t < 16; ++t) o[t] = (v8f){0,0,0,0,0,0,0,0};
  float mrow[8], lrow[8];
  for (int r = 0; r < 8; ++r) { mrow[r] = -3.0e38f; lrow[r] = 0.0f; }

  const float scale = 0.03125f;  // 1/sqrt(1024)

#if USE_TDM
  // prologue: TDM streams K tile 0 into buffer 0 (one issue per WG, wave 0)
  if (wave == 0)
    tdm_load_1d_f16(Kb, (unsigned)(uintptr_t)Ks0, LKT * GD);
#endif

  for (int k0 = 0; k0 < Lk; k0 += LKT) {
    const half_t* KsC;
#if USE_TDM
    const int bufsel = (k0 >> 5) & 1;
    KsC = bufsel ? Ks1 : Ks0;
    if (wave == 0) {
      // issue next tile into the other buffer, then wait for current tile
      if (k0 + LKT < Lk) {
        tdm_load_1d_f16(Kb + (size_t)(k0 + LKT) * GD,
                        (unsigned)(uintptr_t)(bufsel ? Ks0 : Ks1), LKT * GD);
        __builtin_amdgcn_s_wait_tensorcnt(1);
      } else {
        __builtin_amdgcn_s_wait_tensorcnt(0);
      }
    }
    stage_v_transposed(Vb, Vt, k0, tid);
#else
    KsC = Ks0;
    // manual K copy (B128 -> B128) + V transpose staging
#pragma unroll 4
    for (int j = 0; j < 32; ++j) {
      const int i = tid + j * 128;
      const int r = i >> 7;
      const int c8 = i & 127;
      uint4 kv = *(const uint4*)(Kb + (size_t)(k0 + r) * GD + c8 * 8);
      *(uint4*)(Ks0 + r * GD + c8 * 8) = kv;
    }
    stage_v_transposed(Vb, Vt, k0, tid);
#endif
    __syncthreads();

    // ---- partial S(16x32) over this wave's d-chunk: 16 WMMAs
    v8f s0 = (v8f){0,0,0,0,0,0,0,0}, s1 = (v8f){0,0,0,0,0,0,0,0};
#pragma unroll
    for (int s = 0; s < 8; ++s) {
      const int kb = chunk + s * 32 + (lane >> 4) * 8;
      v16h b0, b1;  // B = K^T: lane n -> K row (tile col), d = kb+{0..7}, kb+16+{0..7}
      ((uint4*)&b0)[0] = *(const uint4*)(KsC + (lane & 15) * GD + kb);
      ((uint4*)&b0)[1] = *(const uint4*)(KsC + (lane & 15) * GD + kb + 16);
      ((uint4*)&b1)[0] = *(const uint4*)(KsC + (16 + (lane & 15)) * GD + kb);
      ((uint4*)&b1)[1] = *(const uint4*)(KsC + (16 + (lane & 15)) * GD + kb + 16);
      s0 = wmma_f16(qf[s], b0, s0);
      s1 = wmma_f16(qf[s], b1, s1);
    }
    // write partial S to LDS (C layout -> row-major [16][32])
    {
      float* sp = Sp + wave * (16 * LKT);
      const int Mh = 8 * (lane >> 4);
#pragma unroll
      for (int r = 0; r < 8; ++r) {
        sp[(r + Mh) * LKT + (lane & 15)]      = s0[r];
        sp[(r + Mh) * LKT + 16 + (lane & 15)] = s1[r];
      }
    }
    __syncthreads();

    // ---- reduce the 4 partials -> full S (identical in every wave),
    // mask + scale + online softmax, emit P (f16) via wave 0.
    const int Mh = 8 * (lane >> 4);
    const int n0c = lane & 15;
    float sf[2][8];
#pragma unroll
    for (int t = 0; t < 2; ++t) {
      const int N = t * 16 + n0c;
      const int mv = maskb[k0 + N];
#pragma unroll
      for (int r = 0; r < 8; ++r) {
        const int idx = (r + Mh) * LKT + N;
        float acc = Sp[idx] + Sp[512 + idx] + Sp[1024 + idx] + Sp[1536 + idx];
        sf[t][r] = mv ? acc * scale : NEG_INF_SCORE;
      }
    }
#pragma unroll
    for (int r = 0; r < 8; ++r) {
      float mx = fmaxf(sf[0][r], sf[1][r]);
      for (int off = 8; off >= 1; off >>= 1)
        mx = fmaxf(mx, __shfl_xor(mx, off));   // reduce within 16-lane half
      const float mnew = fmaxf(mrow[r], mx);
      const float alpha = __expf(mrow[r] - mnew);
      const float e0 = __expf(sf[0][r] - mnew);
      const float e1 = __expf(sf[1][r] - mnew);
      float rs = e0 + e1;
      for (int off = 8; off >= 1; off >>= 1) rs += __shfl_xor(rs, off);
      lrow[r] = lrow[r] * alpha + rs;
      mrow[r] = mnew;
#pragma unroll
      for (int t = 0; t < 16; ++t) o[t][r] *= alpha;  // rescale O
      if (wave == 0) {  // identical in all waves; one writer
        Pb[(r + Mh) * LKT + n0c]      = (half_t)e0;
        Pb[(r + Mh) * LKT + 16 + n0c] = (half_t)e1;
      }
    }
    __syncthreads();

    // ---- O(16x256) += P(16x32) @ V(32x256 chunk): 16 WMMAs
    v16h pf;
    ((uint4*)&pf)[0] = *(const uint4*)(Pb + qm * LKT + qks);
    ((uint4*)&pf)[1] = *(const uint4*)(Pb + qm * LKT + 16 + qks);
#pragma unroll
    for (int t = 0; t < 16; ++t) {
      const int vcol = chunk + t * 16 + (lane & 15);
      const int kb = (lane >> 4) * 8;
      v16h bfrag;
      ((uint4*)&bfrag)[0] = *(const uint4*)(Vt + vcol * LKT + kb);
      ((uint4*)&bfrag)[1] = *(const uint4*)(Vt + vcol * LKT + kb + 16);
      o[t] = wmma_f16(pf, bfrag, o[t]);
    }
    __syncthreads();  // protect Vt/Pb (and K buffers) before next stage
  }

  // ---- epilogue: O / l, f32 store
  const int hsel = lane >> 4;
#pragma unroll
  for (int t = 0; t < 16; ++t) {
    const int col = chunk + t * 16 + (lane & 15);
#pragma unroll
    for (int r = 0; r < 8; ++r) {
      const int row = q0 + r + hsel * 8;
      O[((size_t)b * Lq + row) * GD + col] = o[t][r] / lrow[r];
    }
  }
}

// ---------------------------------------------------------------------------
extern "C" void kernel_launch(void* const* d_in, const int* in_sizes, int n_in,
                              void* d_out, int out_size, void* d_ws, size_t ws_size,
                              hipStream_t stream) {
  (void)in_sizes; (void)n_in; (void)out_size; (void)ws_size;
  const float* hm    = (const float*)d_in[0];
  const float* text  = (const float*)d_in[1];
  const int*   tmask = (const int*)d_in[2];
  const float* Wq = (const float*)d_in[3];
  const float* bq = (const float*)d_in[4];
  const float* Wk = (const float*)d_in[5];
  const float* bk = (const float*)d_in[6];
  const float* Wv = (const float*)d_in[7];
  const float* bv = (const float*)d_in[8];
  float* out = (float*)d_out;

  const int B = 8, Lq = 2048, Lk = 2048;
  const size_t elems = (size_t)B * Lq * GD;  // == B*Lk*GD

  half_t* Qh = (half_t*)d_ws;
  half_t* Kh = Qh + elems;
  half_t* Vh = Kh + elems;

  dim3 gproj((B * Lq) / TM, GD / TN);
  proj_gemm_f16<<<gproj, 128, 0, stream>>>(hm,   Wq, bq, Qh, B * Lq);
  proj_gemm_f16<<<gproj, 128, 0, stream>>>(text, Wk, bk, Kh, B * Lk);
  proj_gemm_f16<<<gproj, 128, 0, stream>>>(text, Wv, bv, Vh, B * Lk);

  const size_t smem = (size_t)3 * LKT * GD * 2   // Ks0 + Ks1 + Vt
                    + (size_t)4 * 16 * LKT * 4   // Sp
                    + (size_t)16 * LKT * 2;      // Pb
  flash_attn<<<dim3(Lq / 16, B), 128, smem, stream>>>(Qh, Kh, Vh, tmask, out,
                                                      B, Lq, Lk);
}
